// GraphAutoEncoder_50130858278913
// MI455X (gfx1250) — matrix operand order repaired and checked by reading, not compiled, not run on hardware
//
#include <hip/hip_runtime.h>

#define DIM_D 3072
#define DIM_H 1536
#define DIM_L 64
#define LN_EPS 1e-5f

typedef __attribute__((ext_vector_type(16))) __bf16 v16bf;
typedef __attribute__((ext_vector_type(8)))  __bf16 v8bf;
typedef __attribute__((ext_vector_type(8)))  float  v8f;

// fp32 -> bf16 round-to-nearest-even, kept in 32-bit registers.
static __device__ __forceinline__ unsigned bf16rne(float f) {
    unsigned u = __builtin_bit_cast(unsigned, f);
    return u + 0x7FFFu + ((u >> 16) & 1u);           // bf16 lives in high 16 bits
}
// pack two bf16 (lo, hi) with one v_perm_b32
static __device__ __forceinline__ unsigned pack_bf16x2(float lo, float hi) {
    return __builtin_amdgcn_perm(bf16rne(hi), bf16rne(lo), 0x07060302u);
}
static __device__ __forceinline__ unsigned short bf16h(float f) {
    return (unsigned short)(bf16rne(f) >> 16);
}

// ---------------------------------------------------------------------------
// Degree / normalization kernels
// ---------------------------------------------------------------------------
__global__ void init_deg_kernel(float* deg, int n) {
    int i = blockIdx.x * blockDim.x + threadIdx.x;
    if (i < n) deg[i] = 1.0f;                          // self-loop
}
__global__ void count_deg_kernel(float* deg, const long long* ei, int e) {
    int i = blockIdx.x * blockDim.x + threadIdx.x;
    if (i < e) atomicAdd(&deg[ei[(long)e + i]], 1.0f); // dst row of edge_index
}
__global__ void finalize_dinv_kernel(float* deg, int n) {
    int i = blockIdx.x * blockDim.x + threadIdx.x;
    if (i < n) deg[i] = rsqrtf(fmaxf(deg[i], 1e-12f));
}

// ---------------------------------------------------------------------------
// bf16 WMMA GEMM:  C[M,NN] = act(A[M,K] @ B[K,NN] + bias)
// 256 threads = 8 waves arranged MWAVES x NWAVES; wave tile (AF*16) x (BF*16).
// Block tile BM x BN, K step 32, double-buffered LDS (1 barrier / K-step).
// flags: bit0 = add bias, bit1 = relu.  Requires K%32==0, NN%BN==0.
// M handled with clamp (stage) + guard (store).
// ---------------------------------------------------------------------------
template <int MWAVES, int NWAVES, int AF, int BF>
__global__ __launch_bounds__(256) void wmma_gemm_kernel(
    const float* __restrict__ A, const float* __restrict__ B,
    const float* __restrict__ bias, float* __restrict__ C,
    int M, int K, int NN, int flags)
{
    constexpr int BM = MWAVES * AF * 16;
    constexpr int BN = NWAVES * BF * 16;
    constexpr int AV = (BM * 32) / 1024;      // float4 loads / thread for A tile
    constexpr int BV = (BN * 32) / 1024;      // float4 loads / thread for B tile
    constexpr int BROW = BN / 4;              // threads covering one B k-row
    constexpr int KST = 256 / BROW;           // k rows per staging pass

    __shared__ __bf16 ldsA[2][BM * 32];       // [m][k]
    __shared__ __bf16 ldsBt[2][BN * 32];      // [n][k]  (B transposed)

    const int tid  = threadIdx.x;
    const int lane = tid & 31;
    const int wave = tid >> 5;
    const int waveM = wave % MWAVES;
    const int waveN = wave / MWAVES;
    const long m0 = (long)blockIdx.x * BM;
    const long n0 = (long)blockIdx.y * BN;

    // WMMA wave32 fragment geometry (ISA 7.12.2)
    const int rowA = lane & 15;
    const int kbA  = (lane < 16) ? 0 : 16;    // A: K = kb..kb+15
    const int kbB  = (lane < 16) ? 0 : 16;    // B: K = kb..kb+15
    const int ln   = lane & 15;

    // staging coords
    const int amr = tid >> 3;                 // A row base (0..31)
    const int akc = (tid & 7) << 2;           // A col (float4)
    const int bk  = tid / BROW;               // B k-row base
    const int bn4 = (tid % BROW) << 2;        // B col (float4)

    const float* Ap[AV];
#pragma unroll
    for (int i = 0; i < AV; ++i) {
        long gm = m0 + amr + 32 * i;
        if (gm >= M) gm = M - 1;              // clamp (stores are guarded)
        Ap[i] = A + gm * (long)K + akc;
    }
    const float* Bp[BV];
#pragma unroll
    for (int i = 0; i < BV; ++i)
        Bp[i] = B + (long)(bk + KST * i) * NN + n0 + bn4;

    v8f acc[AF][BF];
#pragma unroll
    for (int a = 0; a < AF; ++a)
#pragma unroll
        for (int b = 0; b < BF; ++b) { v8f z = {}; acc[a][b] = z; }

    float4 ar[AV], br[BV];

    auto load_next = [&]() {
#pragma unroll
        for (int i = 0; i < AV; ++i) { ar[i] = *(const float4*)Ap[i]; Ap[i] += 32; }
#pragma unroll
        for (int i = 0; i < BV; ++i) { br[i] = *(const float4*)Bp[i]; Bp[i] += (long)32 * NN; }
    };
    auto stage = [&](int pb) {
#pragma unroll
        for (int i = 0; i < AV; ++i) {
            uint2 u;
            u.x = pack_bf16x2(ar[i].x, ar[i].y);
            u.y = pack_bf16x2(ar[i].z, ar[i].w);
            *(uint2*)&ldsA[pb][(amr + 32 * i) * 32 + akc] = u;
        }
        unsigned short* bt = (unsigned short*)ldsBt[pb];
#pragma unroll
        for (int i = 0; i < BV; ++i) {
            const int kk = bk + KST * i;
            bt[(bn4 + 0) * 32 + kk] = bf16h(br[i].x);
            bt[(bn4 + 1) * 32 + kk] = bf16h(br[i].y);
            bt[(bn4 + 2) * 32 + kk] = bf16h(br[i].z);
            bt[(bn4 + 3) * 32 + kk] = bf16h(br[i].w);
        }
    };

    load_next();
    stage(0);
    int p = 0;

    for (int k0 = 0; k0 < K; k0 += 32) {
        __syncthreads();
        const bool nxt = (k0 + 32) < K;
        if (nxt) {
            load_next();                                   // overlap with WMMAs
            __builtin_prefetch(Ap[0], 0, 0);               // pull tile after next
        }

        v16bf af[AF];
#pragma unroll
        for (int a = 0; a < AF; ++a) {
            const __bf16* ap = &ldsA[p][((waveM * AF + a) * 16 + rowA) * 32 + kbA];
            v8bf lo = *(const v8bf*)(ap);
            v8bf hi = *(const v8bf*)(ap + 8);
            af[a] = __builtin_shufflevector(lo, hi,
                0, 1, 2, 3, 4, 5, 6, 7, 8, 9, 10, 11, 12, 13, 14, 15);
        }
#pragma unroll
        for (int b = 0; b < BF; ++b) {
            v16bf bf = *(const v16bf*)&ldsBt[p][((waveN * BF + b) * 16 + ln) * 32 + kbB];
#pragma unroll
            for (int a = 0; a < AF; ++a)
                acc[a][b] = __builtin_amdgcn_wmma_f32_16x16x32_bf16(
                    false, af[a], false, bf, (short)0, acc[a][b], false, false);
        }
        if (nxt) stage(p ^ 1);
        p ^= 1;
    }

    // epilogue: lanes 0-15 -> M=r, lanes 16-31 -> M=8+r
#pragma unroll
    for (int b = 0; b < BF; ++b) {
        const long n_g = n0 + (waveN * BF + b) * 16 + ln;
        const float bv = (flags & 1) ? bias[n_g] : 0.0f;
#pragma unroll
        for (int a = 0; a < AF; ++a) {
            const long mb = m0 + (waveM * AF + a) * 16 + ((lane < 16) ? 0 : 8);
            float* cp = C + mb * (long)NN + n_g;
#pragma unroll
            for (int r = 0; r < 8; ++r) {
                if (mb + r < M) {
                    float v = acc[a][b][r] + bv;
                    if (flags & 2) v = fmaxf(v, 0.0f);
                    cp[(long)r * NN] = v;
                }
            }
        }
    }
}

// ---------------------------------------------------------------------------
// GCN aggregation: out[i,:] = h[i,:]*dinv[i]^2 + bias   (self loop + bias)
// then per edge:   out[dst,:] += h[src,:] * dinv[src]*dinv[dst]
// ---------------------------------------------------------------------------
__global__ void agg_init_kernel(const float* __restrict__ h, float* __restrict__ out,
                                const float* __restrict__ bias, const float* __restrict__ dinv,
                                long total, int C) {
    long i = (long)blockIdx.x * blockDim.x + threadIdx.x;
    if (i < total) {
        long row = i / C;
        int  c   = (int)(i - row * C);
        float w  = dinv[row];
        out[i] = h[i] * w * w + bias[c];
    }
}

__global__ __launch_bounds__(256) void agg_edges_kernel(
    const float* __restrict__ h, float* __restrict__ out,
    const long long* __restrict__ ei, const float* __restrict__ dinv,
    int e, int C) {
    int edge = blockIdx.x * 8 + (threadIdx.x >> 5);
    if (edge >= e) return;
    int lane = threadIdx.x & 31;
    long long s = ei[edge];
    long long d = ei[(long)e + edge];
    float w = dinv[s] * dinv[d];
    const float* hs = h + (long)s * C;
    float* od = out + (long)d * C;
    for (int c = lane; c < C; c += 32)
        atomicAdd(&od[c], hs[c] * w);
}

// ---------------------------------------------------------------------------
// Fused ReLU + LayerNorm over rows of length C (one block per row)
// ---------------------------------------------------------------------------
__global__ void relu_ln_kernel(const float* __restrict__ in, float* __restrict__ out,
                               const float* __restrict__ g, const float* __restrict__ be,
                               int C) {
    __shared__ float s1[256];
    __shared__ float s2[256];
    const long row = blockIdx.x;
    const float* x = in + row * (long)C;
    float sum = 0.f, sq = 0.f;
    for (int c = threadIdx.x; c < C; c += blockDim.x) {
        float v = fmaxf(x[c], 0.0f);
        sum += v; sq += v * v;
    }
    s1[threadIdx.x] = sum; s2[threadIdx.x] = sq;
    __syncthreads();
    for (int off = blockDim.x >> 1; off > 0; off >>= 1) {
        if (threadIdx.x < (unsigned)off) {
            s1[threadIdx.x] += s1[threadIdx.x + off];
            s2[threadIdx.x] += s2[threadIdx.x + off];
        }
        __syncthreads();
    }
    const float mean = s1[0] / C;
    const float var  = s2[0] / C - mean * mean;
    const float rs   = rsqrtf(var + LN_EPS);
    float* o = out + row * (long)C;
    for (int c = threadIdx.x; c < C; c += blockDim.x) {
        float v = fmaxf(x[c], 0.0f);
        o[c] = (v - mean) * rs * g[c] + be[c];
    }
}

// ---------------------------------------------------------------------------
extern "C" void kernel_launch(void* const* d_in, const int* in_sizes, int n_in,
                              void* d_out, int out_size, void* d_ws, size_t ws_size,
                              hipStream_t stream) {
    const float*     x     = (const float*)d_in[0];
    const long long* ei    = (const long long*)d_in[1];   // int64 [2,E]
    const float*     W_gc1 = (const float*)d_in[2];
    const float*     b_gc1 = (const float*)d_in[3];
    const float*     W_gc2 = (const float*)d_in[4];
    const float*     b_gc2 = (const float*)d_in[5];
    const float*     g1    = (const float*)d_in[6];
    const float*     be1   = (const float*)d_in[7];
    const float*     g2    = (const float*)d_in[8];
    const float*     be2   = (const float*)d_in[9];
    const float*     W_fc1 = (const float*)d_in[10];
    const float*     b_fc1 = (const float*)d_in[11];
    const float*     W_fc2 = (const float*)d_in[12];
    const float*     b_fc2 = (const float*)d_in[13];

    const int N = in_sizes[0] / DIM_D;                    // 20000
    const int E = in_sizes[1] / 2;                        // 320000

    char* ws = (char*)d_ws;
    auto carve = [&](size_t bytes) {
        char* p = ws;
        ws += (bytes + 255) & ~((size_t)255);
        return p;
    };
    float* dinv = (float*)carve((size_t)N * sizeof(float));
    float* h1   = (float*)carve((size_t)N * DIM_H * sizeof(float)); // feats / hidden
    float* buf1 = (float*)carve((size_t)N * DIM_H * sizeof(float)); // agg1 / decode
    float* h2   = (float*)carve((size_t)N * DIM_L * sizeof(float));
    float* agg2 = (float*)carve((size_t)N * DIM_L * sizeof(float));

    float* latent = (float*)d_out;                         // [N, L]
    float* recon  = (float*)d_out + (size_t)N * DIM_L;     // [N, D]

    // 1) symmetric GCN normalization
    init_deg_kernel<<<(N + 255) / 256, 256, 0, stream>>>(dinv, N);
    count_deg_kernel<<<(E + 255) / 256, 256, 0, stream>>>(dinv, ei, E);
    finalize_dinv_kernel<<<(N + 255) / 256, 256, 0, stream>>>(dinv, N);

    // 2) GCN layer 1: h1 = x @ W_gc1 ; aggregate ; ReLU+LN -> h1
    {
        dim3 g((N + 127) / 128, DIM_H / 128);
        wmma_gemm_kernel<4, 2, 2, 4><<<g, 256, 0, stream>>>(x, W_gc1, nullptr, h1,
                                                            N, DIM_D, DIM_H, 0);
        long tot = (long)N * DIM_H;
        agg_init_kernel<<<(tot + 255) / 256, 256, 0, stream>>>(h1, buf1, b_gc1, dinv, tot, DIM_H);
        agg_edges_kernel<<<(E + 7) / 8, 256, 0, stream>>>(h1, buf1, ei, dinv, E, DIM_H);
        relu_ln_kernel<<<N, 256, 0, stream>>>(buf1, h1, g1, be1, DIM_H);
    }

    // 3) GCN layer 2: h2 = h1 @ W_gc2 ; aggregate ; ReLU+LN -> latent (d_out)
    {
        dim3 g((N + 63) / 64, DIM_L / 64);
        wmma_gemm_kernel<2, 4, 2, 1><<<g, 256, 0, stream>>>(h1, W_gc2, nullptr, h2,
                                                            N, DIM_H, DIM_L, 0);
        long tot = (long)N * DIM_L;
        agg_init_kernel<<<(tot + 255) / 256, 256, 0, stream>>>(h2, agg2, b_gc2, dinv, tot, DIM_L);
        agg_edges_kernel<<<(E + 7) / 8, 256, 0, stream>>>(h2, agg2, ei, dinv, E, DIM_L);
        relu_ln_kernel<<<N, 64, 0, stream>>>(agg2, latent, g2, be2, DIM_L);
    }

    // 4) decoder: decode = relu(latent @ W_fc1 + b_fc1) -> buf1
    {
        dim3 g((N + 127) / 128, DIM_H / 128);
        wmma_gemm_kernel<4, 2, 2, 4><<<g, 256, 0, stream>>>(latent, W_fc1, b_fc1, buf1,
                                                            N, DIM_L, DIM_H, 3);
    }
    // 5) reconstructed = decode @ W_fc2 + b_fc2 -> d_out tail
    {
        dim3 g((N + 127) / 128, DIM_D / 128);
        wmma_gemm_kernel<4, 2, 2, 4><<<g, 256, 0, stream>>>(buf1, W_fc2, b_fc2, recon,
                                                            N, DIM_H, DIM_D, 1);
    }
}
